// ForeBackGenerator_2370821947516
// MI455X (gfx1250) — matrix operand order repaired and checked by reading, not compiled, not run on hardware
//
#include <hip/hip_runtime.h>
#include <hip/hip_bf16.h>
#include <stdint.h>

#define NIMG 10
#define CCH  512
#define HWD  1600
#define WDT  40
#define OC   128
#define EPSF 1e-12f

typedef __attribute__((ext_vector_type(16))) __bf16 v16bf;
typedef __attribute__((ext_vector_type(8)))  __bf16 v8bf;
typedef __attribute__((ext_vector_type(8)))  float  v8f;

// A-fragment (16x32 bf16): lane m = lane&15, kb = (lane&16)?8:0,
// K-chunks [kb, kb+7] and [kb+16, kb+23] -> two 16B loads.
__device__ __forceinline__ v16bf frag_load_A(const __bf16* __restrict__ base, uint32_t off, int lane) {
  const __bf16* p = base + off + ((lane & 16) ? 8u : 0u);
  v8bf lo = *(const v8bf*)p;
  v8bf hi = *(const v8bf*)(p + 16);
  v16bf r;
#pragma unroll
  for (int i = 0; i < 8; ++i) { r[i] = lo[i]; r[i + 8] = hi[i]; }
  return r;
}
// B-fragment (32x16 bf16): lane n = lane&15, kb = (lane&16)?8:0,
// K-chunks [kb, kb+7] and [kb+16, kb+23] from an NxK row-major array.
__device__ __forceinline__ v16bf frag_load_B(const __bf16* __restrict__ base, uint32_t off, int lane) {
  const __bf16* p = base + off + ((lane & 16) ? 8u : 0u);
  v8bf lo = *(const v8bf*)p;
  v8bf hi = *(const v8bf*)(p + 16);
  v16bf r;
#pragma unroll
  for (int i = 0; i < 8; ++i) { r[i] = lo[i]; r[i + 8] = hi[i]; }
  return r;
}
#define WMMA_BF16(A, B, Cc) \
  __builtin_amdgcn_wmma_f32_16x16x32_bf16(false, (A), false, (B), (short)0, (Cc), false, false)

// ---------------- K1: per-pixel L2 norm + transposed bf16 features ----------------
__global__ void k_norm_transpose(const float* __restrict__ feats, float* __restrict__ invn,
                                 __bf16* __restrict__ T) {
  int i = blockIdx.x * blockDim.x + threadIdx.x;
  int n = blockIdx.y;
  if (i >= HWD) return;
  const float* f = feats + (size_t)n * CCH * HWD + i;
  float ss = 0.f;
  for (int c = 0; c < CCH; ++c) { float v = f[(size_t)c * HWD]; ss += v * v; }
  float inv = 1.f / fmaxf(sqrtf(ss), EPSF);
  invn[n * HWD + i] = inv;
  __bf16* t = T + ((size_t)n * HWD + i) * CCH;
  for (int c = 0; c < CCH; ++c) t[c] = (__bf16)(f[(size_t)c * HWD] * inv);
}

// ---------------- K2: SIV raw = mean_i NF * SISM ----------------
__global__ void k_siv_raw(const float* __restrict__ feats, const float* __restrict__ SISM,
                          const float* __restrict__ invn, float* __restrict__ sivraw) {
  int c = blockIdx.x, n = blockIdx.y, tid = threadIdx.x;
  __shared__ float red[256];
  const float* f = feats + ((size_t)n * CCH + c) * HWD;
  const float* s = SISM + (size_t)n * HWD;
  const float* iv = invn + n * HWD;
  float acc = 0.f;
  for (int i = tid; i < HWD; i += 256) acc += f[i] * iv[i] * s[i];
  red[tid] = acc; __syncthreads();
  for (int st = 128; st > 0; st >>= 1) { if (tid < st) red[tid] += red[tid + st]; __syncthreads(); }
  if (tid == 0) sivraw[n * CCH + c] = red[0] * (1.f / (float)HWD);
}

// ---------------- K3: L2-normalize SIVs over channels ----------------
__global__ void k_siv_norm(const float* __restrict__ sivraw, float* __restrict__ siv) {
  int n = blockIdx.x, tid = threadIdx.x;
  __shared__ float red[256];
  const float* r = sivraw + n * CCH;
  float acc = 0.f;
  for (int c = tid; c < CCH; c += 256) { float v = r[c]; acc += v * v; }
  red[tid] = acc; __syncthreads();
  for (int st = 128; st > 0; st >>= 1) { if (tid < st) red[tid] += red[tid + st]; __syncthreads(); }
  float inv = 1.f / fmaxf(sqrtf(red[0]), EPSF);
  for (int c = tid; c < CCH; c += 256) siv[n * CCH + c] = r[c] * inv;
}

// ---------------- K4: cm[n,m,i] = <NF(n,:,i), SIV(m,:)> (fp32, precision-critical) ----------------
__global__ void k_cm(const float* __restrict__ feats, const float* __restrict__ invn,
                     const float* __restrict__ siv, float* __restrict__ cm) {
  int i = blockIdx.x * blockDim.x + threadIdx.x;
  int n = blockIdx.y;
  if (i >= HWD) return;
  const float* f = feats + (size_t)n * CCH * HWD + i;
  float acc[NIMG];
#pragma unroll
  for (int m = 0; m < NIMG; ++m) acc[m] = 0.f;
  for (int c = 0; c < CCH; ++c) {
    float v = f[(size_t)c * HWD];
#pragma unroll
    for (int m = 0; m < NIMG; ++m) acc[m] += v * siv[m * CCH + c];
  }
  float inv = invn[n * HWD + i];
  for (int m = 0; m < NIMG; ++m) cm[((size_t)(n * NIMG + m)) * HWD + i] = acc[m] * inv;
}

// ---------------- K5: reciprocal L2 norm of each cm row ----------------
__global__ void k_cm_norm(const float* __restrict__ cm, float* __restrict__ rn) {
  int nm = blockIdx.x, tid = threadIdx.x;
  __shared__ float red[256];
  const float* r = cm + (size_t)nm * HWD;
  float acc = 0.f;
  for (int i = tid; i < HWD; i += 256) { float v = r[i]; acc += v * v; }
  red[tid] = acc; __syncthreads();
  for (int st = 128; st > 0; st >>= 1) { if (tid < st) red[tid] += red[tid + st]; __syncthreads(); }
  if (tid == 0) rn[nm] = 1.f / fmaxf(sqrtf(red[0]), EPSF);
}

// ---------------- K6: corr-sums -> softmax -> CSA -> min/max normalize (one block per n) ----------------
__global__ void k_csa(const float* __restrict__ cm, const float* __restrict__ rn,
                      float* __restrict__ CSA) {
  int n = blockIdx.x, tid = threadIdx.x;
  __shared__ float sS[NIMG], swv[NIMG];
  __shared__ float smin[256], smax[256];
  if (tid < NIMG) sS[tid] = 0.f;
  __syncthreads();
  float r[NIMG];
#pragma unroll
  for (int m = 0; m < NIMG; ++m) r[m] = rn[n * NIMG + m];
  float localS[NIMG];
#pragma unroll
  for (int m = 0; m < NIMG; ++m) localS[m] = 0.f;
  // S[m] = sum_i cmn[m,i] * (sum_p cmn[p,i])  == sum_p corr[m,p]
  for (int i = tid; i < HWD; i += 256) {
    float cv[NIMG]; float s = 0.f;
#pragma unroll
    for (int m = 0; m < NIMG; ++m) { cv[m] = cm[((size_t)(n * NIMG + m)) * HWD + i] * r[m]; s += cv[m]; }
#pragma unroll
    for (int m = 0; m < NIMG; ++m) localS[m] += cv[m] * s;
  }
#pragma unroll
  for (int m = 0; m < NIMG; ++m) atomicAdd(&sS[m], localS[m]);
  __syncthreads();
  if (tid == 0) {
    float mx = sS[0];
    for (int m = 1; m < NIMG; ++m) mx = fmaxf(mx, sS[m]);
    float sum = 0.f;
    for (int m = 0; m < NIMG; ++m) { float e = expf(sS[m] - mx); swv[m] = e; sum += e; }
    for (int m = 0; m < NIMG; ++m) swv[m] /= sum;
  }
  __syncthreads();
  float myv[7]; int cnt = 0;
  float lmin = 3.4e38f, lmax = -3.4e38f;
  for (int i = tid; i < HWD; i += 256) {
    float csa = 0.f;
#pragma unroll
    for (int m = 0; m < NIMG; ++m) csa += cm[((size_t)(n * NIMG + m)) * HWD + i] * r[m] * swv[m];
    myv[cnt++] = csa;
    lmin = fminf(lmin, csa); lmax = fmaxf(lmax, csa);
  }
  smin[tid] = lmin; smax[tid] = lmax; __syncthreads();
  for (int st = 128; st > 0; st >>= 1) {
    if (tid < st) { smin[tid] = fminf(smin[tid], smin[tid + st]); smax[tid] = fmaxf(smax[tid], smax[tid + st]); }
    __syncthreads();
  }
  float mn = smin[0];
  float scale = 1.f / (smax[0] - mn + EPSF);
  cnt = 0;
  for (int i = tid; i < HWD; i += 256) CSA[n * HWD + i] = (myv[cnt++] - mn) * scale;
}

// ---------------- K7: stable descending argsort via O(L^2) rank ----------------
__global__ void k_rank(const float* __restrict__ CSA, int* __restrict__ idx) {
  int n = blockIdx.x, tid = threadIdx.x;
  __shared__ float v[HWD];
  for (int i = tid; i < HWD; i += 256) v[i] = CSA[n * HWD + i];
  __syncthreads();
  for (int p = tid; p < HWD; p += 256) {
    float vp = v[p]; int rank = 0;
    for (int j = 0; j < HWD; ++j) {
      float vj = v[j];
      rank += (vj > vp) || (vj == vp && j < p);
    }
    idx[n * HWD + rank] = p;
  }
}

// ---------------- weight prep: taps-major bf16 layouts ----------------
__global__ void k_prep_w1(const float* __restrict__ w1, __bf16* __restrict__ w1t) {
  int e = blockIdx.x * 256 + threadIdx.x;
  if (e >= OC * HWD * 9) return;
  int t = e % 9, p = (e / 9) % HWD, o = e / (9 * HWD);
  w1t[((size_t)t * OC + o) * HWD + p] = (__bf16)w1[e];
}
__global__ void k_prep_w2(const float* __restrict__ w2, __bf16* __restrict__ w2t) {
  int e = blockIdx.x * 256 + threadIdx.x;
  if (e >= OC * OC * 9) return;
  int t = e % 9, c = (e / 9) % OC, o = e / (9 * OC);
  w2t[((size_t)t * OC + o) * OC + c] = (__bf16)w2[e];
}
__global__ void k_prep_ws(const float* __restrict__ ws, __bf16* __restrict__ wst) {
  int e = blockIdx.x * 256 + threadIdx.x;
  if (e >= OC * HWD) return;
  wst[e] = (__bf16)ws[e];
}
// zero-pad rows (row NIMG*HWD of xT and hT) used for out-of-image conv taps
__global__ void k_zero_pad(__bf16* __restrict__ xTz, __bf16* __restrict__ hTz) {
  int i = blockIdx.x * 256 + threadIdx.x;
  if (i < HWD) xTz[i] = (__bf16)0.0f;
  if (i < OC)  hTz[i] = (__bf16)0.0f;
}

// ---------------- K8: WMMA Gram + channel gather + CSA scale -> xT[n][j][p] bf16 ----------------
// block: 8 waves; wave = 32 rows x 64 cols; block tile = 256 rows x 64 cols
__global__ void k_gram_xT(const __bf16* __restrict__ T, const int* __restrict__ idx,
                          const float* __restrict__ CSA, __bf16* __restrict__ xT) {
  int tm = blockIdx.x, tn = blockIdx.y, n = blockIdx.z;
  int tid = threadIdx.x, lane = tid & 31, w = tid >> 5;
  int ln15 = lane & 15;
  int m0 = tm * 256 + w * 32;
  int j0 = tn * 64;
  uint32_t aOff[2];
#pragma unroll
  for (int r = 0; r < 2; ++r) {
    int gm = m0 + r * 16 + ln15;
    int gmc = gm < HWD ? gm : (HWD - 1);
    int src = idx[n * HWD + gmc];
    aOff[r] = (uint32_t)(n * HWD + src) * CCH;
  }
  uint32_t bOff[4];
  int jl[4];
#pragma unroll
  for (int s = 0; s < 4; ++s) {
    jl[s] = j0 + s * 16 + ln15;
    bOff[s] = (uint32_t)(n * HWD + jl[s]) * CCH;
  }
  v8f acc[2][4];
#pragma unroll
  for (int r = 0; r < 2; ++r)
#pragma unroll
    for (int s = 0; s < 4; ++s)
#pragma unroll
      for (int i = 0; i < 8; ++i) acc[r][s][i] = 0.f;

  for (int kk = 0; kk < CCH; kk += 32) {
    v16bf a0 = frag_load_A(T, aOff[0] + kk, lane);
    v16bf a1 = frag_load_A(T, aOff[1] + kk, lane);
    v16bf b[4];
#pragma unroll
    for (int s = 0; s < 4; ++s) b[s] = frag_load_B(T, bOff[s] + kk, lane);
#pragma unroll
    for (int s = 0; s < 4; ++s) acc[0][s] = WMMA_BF16(a0, b[s], acc[0][s]);
#pragma unroll
    for (int s = 0; s < 4; ++s) acc[1][s] = WMMA_BF16(a1, b[s], acc[1][s]);
  }
  int hi8 = (lane & 16) ? 8 : 0;
#pragma unroll
  for (int r = 0; r < 2; ++r) {
    int pb = m0 + r * 16 + hi8;
    if (pb < HWD) {
#pragma unroll
      for (int s = 0; s < 4; ++s) {
        float cs = CSA[n * HWD + jl[s]];
        v8bf o;
#pragma unroll
        for (int v = 0; v < 8; ++v) o[v] = (__bf16)(acc[r][s][v] * cs);
        *(v8bf*)(xT + ((size_t)n * HWD + jl[s]) * HWD + pb) = o;
      }
    }
  }
}

// ---------------- K9: conv1 (3x3, 1600->128) implicit GEMM, +bias, relu -> hT bf16 ----------------
// block tile: 128 out-ch x 160 cols; wave = 32 out-ch x 80 cols (w = wm*1 + wq*4)
__global__ void k_conv1(const __bf16* __restrict__ w1t, const __bf16* __restrict__ xT,
                        const float* __restrict__ b1, __bf16* __restrict__ hT) {
  int tq = blockIdx.x, n = blockIdx.y;
  int tid = threadIdx.x, lane = tid & 31, w = tid >> 5;
  int ln15 = lane & 15;
  int wm = w & 3, wq = w >> 2;
  int o0 = wm * 32;
  int qb = tq * 160 + wq * 80;
  int q[5];
#pragma unroll
  for (int s = 0; s < 5; ++s) q[s] = qb + s * 16 + ln15;
  v8f acc[2][5];
#pragma unroll
  for (int r = 0; r < 2; ++r)
#pragma unroll
    for (int s = 0; s < 5; ++s)
#pragma unroll
      for (int i = 0; i < 8; ++i) acc[r][s][i] = 0.f;
  const uint32_t ZROW = (uint32_t)(NIMG * HWD) * HWD;  // zero-pad row in xT

  for (int t = 0; t < 9; ++t) {
    int dy = t / 3 - 1, dx = t % 3 - 1;
    uint32_t aOff[2];
#pragma unroll
    for (int r = 0; r < 2; ++r)
      aOff[r] = (uint32_t)((t * OC) + o0 + r * 16 + ln15) * HWD;
    uint32_t bOff[5];
#pragma unroll
    for (int s = 0; s < 5; ++s) {
      int qy = q[s] / WDT, qx = q[s] % WDT;
      int ny = qy + dy, nx = qx + dx;
      bool val = ((unsigned)ny < WDT) && ((unsigned)nx < WDT);
      bOff[s] = val ? (uint32_t)(n * HWD + ny * WDT + nx) * HWD : ZROW;
    }
    for (int kk = 0; kk < HWD; kk += 32) {
      v16bf a0 = frag_load_A(w1t, aOff[0] + kk, lane);
      v16bf a1 = frag_load_A(w1t, aOff[1] + kk, lane);
      v16bf b[5];
#pragma unroll
      for (int s = 0; s < 5; ++s) b[s] = frag_load_B(xT, bOff[s] + kk, lane);
#pragma unroll
      for (int s = 0; s < 5; ++s) acc[0][s] = WMMA_BF16(a0, b[s], acc[0][s]);
#pragma unroll
      for (int s = 0; s < 5; ++s) acc[1][s] = WMMA_BF16(a1, b[s], acc[1][s]);
    }
  }
  int hi8 = (lane & 16) ? 8 : 0;
#pragma unroll
  for (int r = 0; r < 2; ++r) {
    int ob = o0 + r * 16 + hi8;
#pragma unroll
    for (int s = 0; s < 5; ++s) {
      v8bf o;
#pragma unroll
      for (int v = 0; v < 8; ++v) o[v] = (__bf16)fmaxf(acc[r][s][v] + b1[ob + v], 0.f);
      *(v8bf*)(hT + ((size_t)n * HWD + q[s]) * OC + ob) = o;
    }
  }
}

// ---------------- K10: shortcut 1x1 (1600->128) GEMM -> SC fp32 ----------------
__global__ void k_shortcut(const __bf16* __restrict__ wst, const __bf16* __restrict__ xT,
                           const float* __restrict__ bs, float* __restrict__ SC) {
  int tq = blockIdx.x, n = blockIdx.y;
  int tid = threadIdx.x, lane = tid & 31, w = tid >> 5;
  int ln15 = lane & 15;
  int wm = w & 3, wq = w >> 2;
  int o0 = wm * 32;
  int qb = tq * 160 + wq * 80;
  uint32_t aOff[2];
#pragma unroll
  for (int r = 0; r < 2; ++r) aOff[r] = (uint32_t)(o0 + r * 16 + ln15) * HWD;
  int q[5]; uint32_t bOff[5];
#pragma unroll
  for (int s = 0; s < 5; ++s) {
    q[s] = qb + s * 16 + ln15;
    bOff[s] = (uint32_t)(n * HWD + q[s]) * HWD;
  }
  v8f acc[2][5];
#pragma unroll
  for (int r = 0; r < 2; ++r)
#pragma unroll
    for (int s = 0; s < 5; ++s)
#pragma unroll
      for (int i = 0; i < 8; ++i) acc[r][s][i] = 0.f;
  for (int kk = 0; kk < HWD; kk += 32) {
    v16bf a0 = frag_load_A(wst, aOff[0] + kk, lane);
    v16bf a1 = frag_load_A(wst, aOff[1] + kk, lane);
    v16bf b[5];
#pragma unroll
    for (int s = 0; s < 5; ++s) b[s] = frag_load_B(xT, bOff[s] + kk, lane);
#pragma unroll
    for (int s = 0; s < 5; ++s) acc[0][s] = WMMA_BF16(a0, b[s], acc[0][s]);
#pragma unroll
    for (int s = 0; s < 5; ++s) acc[1][s] = WMMA_BF16(a1, b[s], acc[1][s]);
  }
  int hi8 = (lane & 16) ? 8 : 0;
#pragma unroll
  for (int r = 0; r < 2; ++r) {
    int ob = o0 + r * 16 + hi8;
#pragma unroll
    for (int s = 0; s < 5; ++s)
#pragma unroll
      for (int v = 0; v < 8; ++v)
        SC[((size_t)n * OC + ob + v) * HWD + q[s]] = acc[r][s][v] + bs[ob + v];
  }
}

// ---------------- K11: conv2 (3x3, 128->128) + SC + bias, relu -> d_out ----------------
__global__ void k_conv2_out(const __bf16* __restrict__ w2t, const __bf16* __restrict__ hT,
                            const float* __restrict__ b2, const float* __restrict__ SC,
                            float* __restrict__ out) {
  int tq = blockIdx.x, n = blockIdx.y;
  int tid = threadIdx.x, lane = tid & 31, w = tid >> 5;
  int ln15 = lane & 15;
  int wm = w & 3, wq = w >> 2;
  int o0 = wm * 32;
  int qb = tq * 160 + wq * 80;
  int q[5];
#pragma unroll
  for (int s = 0; s < 5; ++s) q[s] = qb + s * 16 + ln15;
  v8f acc[2][5];
#pragma unroll
  for (int r = 0; r < 2; ++r)
#pragma unroll
    for (int s = 0; s < 5; ++s)
#pragma unroll
      for (int i = 0; i < 8; ++i) acc[r][s][i] = 0.f;
  const uint32_t ZROW = (uint32_t)(NIMG * HWD) * OC;  // zero-pad row in hT

  for (int t = 0; t < 9; ++t) {
    int dy = t / 3 - 1, dx = t % 3 - 1;
    uint32_t aOff[2];
#pragma unroll
    for (int r = 0; r < 2; ++r)
      aOff[r] = (uint32_t)((t * OC) + o0 + r * 16 + ln15) * OC;
    uint32_t bOff[5];
#pragma unroll
    for (int s = 0; s < 5; ++s) {
      int qy = q[s] / WDT, qx = q[s] % WDT;
      int ny = qy + dy, nx = qx + dx;
      bool val = ((unsigned)ny < WDT) && ((unsigned)nx < WDT);
      bOff[s] = val ? (uint32_t)(n * HWD + ny * WDT + nx) * OC : ZROW;
    }
#pragma unroll
    for (int kk = 0; kk < OC; kk += 32) {
      v16bf a0 = frag_load_A(w2t, aOff[0] + kk, lane);
      v16bf a1 = frag_load_A(w2t, aOff[1] + kk, lane);
      v16bf b[5];
#pragma unroll
      for (int s = 0; s < 5; ++s) b[s] = frag_load_B(hT, bOff[s] + kk, lane);
#pragma unroll
      for (int s = 0; s < 5; ++s) acc[0][s] = WMMA_BF16(a0, b[s], acc[0][s]);
#pragma unroll
      for (int s = 0; s < 5; ++s) acc[1][s] = WMMA_BF16(a1, b[s], acc[1][s]);
    }
  }
  int hi8 = (lane & 16) ? 8 : 0;
#pragma unroll
  for (int r = 0; r < 2; ++r) {
    int ob = o0 + r * 16 + hi8;
#pragma unroll
    for (int s = 0; s < 5; ++s)
#pragma unroll
      for (int v = 0; v < 8; ++v) {
        size_t a = ((size_t)n * OC + ob + v) * HWD + q[s];
        out[a] = fmaxf(acc[r][s][v] + b2[ob + v] + SC[a], 0.f);
      }
  }
}

extern "C" void kernel_launch(void* const* d_in, const int* in_sizes, int n_in,
                              void* d_out, int out_size, void* d_ws, size_t ws_size,
                              hipStream_t stream) {
  (void)in_sizes; (void)n_in; (void)out_size; (void)ws_size;
  const float* feats = (const float*)d_in[0];
  const float* SISMs = (const float*)d_in[1];
  const float* w1    = (const float*)d_in[2];
  const float* b1    = (const float*)d_in[3];
  const float* w2    = (const float*)d_in[4];
  const float* b2    = (const float*)d_in[5];
  const float* wsp   = (const float*)d_in[6];
  const float* bs    = (const float*)d_in[7];
  float* out = (float*)d_out;

  uint8_t* p = (uint8_t*)d_ws;
  auto alloc = [&](size_t bytes) -> void* {
    void* r = (void*)p;
    p += (bytes + 255) & ~(size_t)255;
    return r;
  };
  float*  invn   = (float*)alloc((size_t)NIMG * HWD * 4);
  __bf16* T      = (__bf16*)alloc((size_t)NIMG * HWD * CCH * 2);
  float*  sivraw = (float*)alloc((size_t)NIMG * CCH * 4);
  float*  siv    = (float*)alloc((size_t)NIMG * CCH * 4);
  float*  cm     = (float*)alloc((size_t)NIMG * NIMG * HWD * 4);
  float*  rn     = (float*)alloc((size_t)NIMG * NIMG * 4);
  float*  CSA    = (float*)alloc((size_t)NIMG * HWD * 4);
  int*    idx    = (int*)alloc((size_t)NIMG * HWD * 4);
  __bf16* xT     = (__bf16*)alloc(((size_t)NIMG * HWD + 1) * HWD * 2);  // +1 zero row
  __bf16* w1t    = (__bf16*)alloc((size_t)9 * OC * HWD * 2);
  __bf16* w2t    = (__bf16*)alloc((size_t)9 * OC * OC * 2);
  __bf16* wst    = (__bf16*)alloc((size_t)OC * HWD * 2);
  __bf16* hT     = (__bf16*)alloc(((size_t)NIMG * HWD + 1) * OC * 2);   // +1 zero row
  float*  SC     = (float*)alloc((size_t)NIMG * OC * HWD * 4);

  dim3 b256(256);
  // scalar chain (fp32, precision-critical)
  k_norm_transpose<<<dim3(7, NIMG), b256, 0, stream>>>(feats, invn, T);
  k_siv_raw<<<dim3(CCH, NIMG), b256, 0, stream>>>(feats, SISMs, invn, sivraw);
  k_siv_norm<<<dim3(NIMG), b256, 0, stream>>>(sivraw, siv);
  k_cm<<<dim3(7, NIMG), b256, 0, stream>>>(feats, invn, siv, cm);
  k_cm_norm<<<dim3(NIMG * NIMG), b256, 0, stream>>>(cm, rn);
  k_csa<<<dim3(NIMG), b256, 0, stream>>>(cm, rn, CSA);
  k_rank<<<dim3(NIMG), b256, 0, stream>>>(CSA, idx);
  // weight conversion + zero-pad rows
  k_prep_w1<<<dim3((OC * HWD * 9 + 255) / 256), b256, 0, stream>>>(w1, w1t);
  k_prep_w2<<<dim3((OC * OC * 9 + 255) / 256), b256, 0, stream>>>(w2, w2t);
  k_prep_ws<<<dim3((OC * HWD + 255) / 256), b256, 0, stream>>>(wsp, wst);
  k_zero_pad<<<dim3(7), b256, 0, stream>>>(xT + (size_t)NIMG * HWD * HWD,
                                           hT + (size_t)NIMG * HWD * OC);
  // WMMA heavy lifting
  k_gram_xT<<<dim3(7, 25, NIMG), b256, 0, stream>>>(T, idx, CSA, xT);
  k_conv1<<<dim3(10, NIMG), b256, 0, stream>>>(w1t, xT, b1, hT);
  k_shortcut<<<dim3(10, NIMG), b256, 0, stream>>>(wst, xT, bs, SC);
  k_conv2_out<<<dim3(10, NIMG), b256, 0, stream>>>(w2t, hT, b2, SC, out);
}
